// GRU_58746562674768
// MI455X (gfx1250) — matrix operand order repaired and checked by reading, not compiled, not run on hardware
//
#include <hip/hip_runtime.h>

// ---------------------------------------------------------------------------
// GRU scan on MI455X (gfx1250): persistent per-batch-tile workgroups, wave32
// WMMA bf16 16x16x32, weights pre-tiled to B-fragment layout in workspace.
// B=256 -> 16 WGs x 16 batch rows; 512 threads = 16 waves; wave owns 64 cols.
// Round 4: launder weight base as a 64-bit SGPR value and rebuild it as an
// explicit address_space(1) pointer so the weight stream lowers to
// global_load_b128 (LOADcnt only) instead of flat_load_b128 (LOADcnt+DScnt,
// LDS-path contention). Anti-LICM laundering retained from round 3.
// ---------------------------------------------------------------------------

#define H      1024
#define E      10
#define KDIM   1034           // H + E
#define KT     33             // k-tiles of 32 -> covers 1056 (padded, zeros)
#define NT     64             // n-tiles of 16 -> 1024 columns
#define BROWS  16             // batch rows per workgroup (WMMA M tile)
#define ROWU   133            // uint4 per LDS activation row (2128 B, 16B aligned,
                              //  532 dwords -> 20*m mod 64: conflict-free b128)
#define ROWH   (ROWU * 8)     // 1064 bf16 elements per row

typedef __attribute__((ext_vector_type(16))) __bf16        v16bf;
typedef __attribute__((ext_vector_type(8)))  float         v8f;
typedef __attribute__((ext_vector_type(4)))  unsigned int  v4u;

#define GLOBAL_AS __attribute__((address_space(1)))
typedef GLOBAL_AS const v4u* gv4u_p;

union Frag { v4u u[2]; v16bf bf; };

__device__ __forceinline__ unsigned short f2bf(float f) {
  unsigned int u = __float_as_uint(f);
  u += 0x7FFFu + ((u >> 16) & 1u);          // round-to-nearest-even
  return (unsigned short)(u >> 16);
}
__device__ __forceinline__ float bf2f(unsigned short s) {
  return __uint_as_float(((unsigned int)s) << 16);
}
__device__ __forceinline__ unsigned int pack2bf(float a, float b) {
  return (unsigned int)f2bf(a) | ((unsigned int)f2bf(b) << 16);
}
__device__ __forceinline__ float sigm(float x) {
  return 1.0f / (1.0f + __expf(-x));
}
// Opaque pass-through (SGPR pair): breaks load-address invariance so weight
// loads cannot be hoisted out of the recurrence loop; result is rebuilt as an
// address_space(1) pointer so loads lower to global_load_* (not flat_*).
__device__ __forceinline__ gv4u_p glaunder(const void* p) {
  unsigned long long u = (unsigned long long)p;
  asm volatile("" : "+s"(u));
  return (gv4u_p)u;
}

// ---------------------------------------------------------------------------
// Prep: fp32 weights -> bf16, tiled into per-lane WMMA B-fragment order.
// flat = (((g*NT + nt)*KT + kt)*512) + lane*16 + i
//   n = nt*16 + (lane & 15)
//   k = kt*32 + ((lane >> 4) << 4) + i     (lane<16: K 0..15, lane>=16: K 16..31)
// ---------------------------------------------------------------------------
__global__ void gru_prep(const float* __restrict__ Wz,
                         const float* __restrict__ Wr,
                         const float* __restrict__ W,
                         unsigned short* __restrict__ wbuf) {
  int gid = blockIdx.x * blockDim.x + threadIdx.x;
  const int total = 3 * NT * KT * 512;
  if (gid >= total) return;
  int i    = gid & 15;
  int lane = (gid >> 4) & 31;
  int kt   = (gid >> 9) % KT;
  int nt   = (gid / (512 * KT)) % NT;
  int g    = gid / (512 * KT * NT);
  int n = nt * 16 + (lane & 15);
  int k = kt * 32 + ((lane >> 4) << 4) + i;
  const float* Wg = (g == 0) ? Wz : (g == 1) ? Wr : W;
  float v = (k < KDIM) ? Wg[k * H + n] : 0.0f;
  wbuf[gid] = f2bf(v);
}

// ---------------------------------------------------------------------------
// Main persistent GRU kernel: one WG per 16-row batch tile, no global sync.
// ---------------------------------------------------------------------------
__global__ __launch_bounds__(512, 1) void gru_main(
    const int*            __restrict__ x,     // [256][512] token ids
    const float*          __restrict__ emb,   // [10][10]
    const unsigned short* __restrict__ wbuf,  // tiled bf16 Wz|Wr|W
    const float*          __restrict__ Wph,   // [1024][10]
    const float*          __restrict__ bp,    // [10]
    float*                __restrict__ out)   // [256][10]
{
  __shared__ v4u cAv[BROWS * ROWU];           // bf16 activation [16][1064]
  unsigned short* cAs = (unsigned short*)cAv;
  const v4u* cAr = (const v4u*)cAv;

  const int tid  = threadIdx.x;
  const int lane = tid & 31;
  const int wave = tid >> 5;                  // 0..15
  const int hi   = lane >> 4;                 // half-wave select
  const int lo   = lane & 15;
  const int wg   = blockIdx.x;                // batch tile

  // zero LDS (h region and K padding must start at 0)
  {
    v4u zero = 0;
    for (int idx = tid; idx < BROWS * ROWU; idx += 512) cAv[idx] = zero;
  }

  float        h_keep[4][8];                  // fp32 hidden state (own columns)
  unsigned int z_pk[4][4];                    // update gate, packed bf16 pairs
  unsigned int t_pk[4][4];                    // r*h staging, packed bf16 pairs
  #pragma unroll
  for (int nt = 0; nt < 4; ++nt)
    #pragma unroll
    for (int v = 0; v < 8; ++v) h_keep[nt][v] = 0.0f;

  __syncthreads();

  const int aBase = lo * ROWU + hi * 2;       // A-frag v4u index: m*133 + kt*4 + hi*2

  for (int t = 0; t < 512; ++t) {
    // ---- stage embedding x_t into cA[.., 1024..1033] (rest stays zero) ----
    if (tid < BROWS) {
      int xid = x[(wg * BROWS + tid) * 512 + t];
      #pragma unroll
      for (int e = 0; e < E; ++e)
        cAs[tid * ROWH + H + e] = f2bf(emb[xid * E + e]);
    }
    __syncthreads();

    // ---- phase 1: z = sigmoid(c@Wz), r = sigmoid(c@Wr) -------------------
    #pragma unroll
    for (int nt = 0; nt < 4; ++nt) {
      const int ntile = wave * 4 + nt;
      gv4u_p wnt = glaunder(wbuf);            // opaque per n-tile, SGPR base
      v8f zacc = {}; v8f racc = {};
      gv4u_p pz = wnt + (0 * NT + ntile) * KT * 64 + lane * 2;
      gv4u_p pr = wnt + (1 * NT + ntile) * KT * 64 + lane * 2;
      int ai = aBase;
      #pragma unroll 2
      for (int kt = 0; kt < KT; ++kt) {
        Frag a, fz, fr;
        a.u[0]  = cAr[ai]; a.u[1]  = cAr[ai + 1];
        fz.u[0] = pz[0];   fz.u[1] = pz[1];
        fr.u[0] = pr[0];   fr.u[1] = pr[1];
        zacc = __builtin_amdgcn_wmma_f32_16x16x32_bf16(
                   false, a.bf, false, fz.bf, (short)0, zacc, false, false);
        racc = __builtin_amdgcn_wmma_f32_16x16x32_bf16(
                   false, a.bf, false, fr.bf, (short)0, racc, false, false);
        ai += 4; pz += 64; pr += 64;
      }
      #pragma unroll
      for (int p = 0; p < 4; ++p) {
        float z0 = sigm(zacc[2 * p]), z1 = sigm(zacc[2 * p + 1]);
        float r0 = sigm(racc[2 * p]), r1 = sigm(racc[2 * p + 1]);
        z_pk[nt][p] = pack2bf(z0, z1);
        t_pk[nt][p] = pack2bf(r0 * h_keep[nt][2 * p], r1 * h_keep[nt][2 * p + 1]);
      }
      asm volatile("" ::: "memory");          // keep nt iterations sequential
    }
    __syncthreads();                          // all cA reads complete

    // ---- overwrite h-region of cA with r*h (bf16) ------------------------
    #pragma unroll
    for (int nt = 0; nt < 4; ++nt) {
      const int n = (wave * 4 + nt) * 16 + lo;
      #pragma unroll
      for (int p = 0; p < 4; ++p) {
        unsigned int u = t_pk[nt][p];
        cAs[(2 * p     + hi * 8) * ROWH + n] = (unsigned short)u;
        cAs[(2 * p + 1 + hi * 8) * ROWH + n] = (unsigned short)(u >> 16);
      }
    }
    __syncthreads();

    // ---- phase 2: h_hat = tanh([r*h, x]@W); h = (1-z)h + z*h_hat ---------
    #pragma unroll
    for (int nt = 0; nt < 4; ++nt) {
      const int ntile = wave * 4 + nt;
      gv4u_p wnt = glaunder(wbuf);            // opaque per n-tile, SGPR base
      v8f hacc = {};
      gv4u_p pw = wnt + (2 * NT + ntile) * KT * 64 + lane * 2;
      int ai = aBase;
      #pragma unroll 2
      for (int kt = 0; kt < KT; ++kt) {
        Frag a, fw;
        a.u[0]  = cAr[ai]; a.u[1]  = cAr[ai + 1];
        fw.u[0] = pw[0];   fw.u[1] = pw[1];
        hacc = __builtin_amdgcn_wmma_f32_16x16x32_bf16(
                   false, a.bf, false, fw.bf, (short)0, hacc, false, false);
        ai += 4; pw += 64;
      }
      #pragma unroll
      for (int p = 0; p < 4; ++p) {
        unsigned int zu = z_pk[nt][p];
        #pragma unroll
        for (int q = 0; q < 2; ++q) {
          int v = 2 * p + q;
          float e2 = __expf(2.0f * hacc[v]);
          float th = 1.0f - 2.0f / (e2 + 1.0f);        // tanh
          float z  = bf2f((unsigned short)(q ? (zu >> 16) : zu));
          h_keep[nt][v] = (1.0f - z) * h_keep[nt][v] + z * th;
        }
      }
      asm volatile("" ::: "memory");          // keep nt iterations sequential
    }
    __syncthreads();                          // all cA reads complete

    // ---- write new h back into cA (bf16) for next step -------------------
    #pragma unroll
    for (int nt = 0; nt < 4; ++nt) {
      const int n = (wave * 4 + nt) * 16 + lo;
      #pragma unroll
      for (int v = 0; v < 8; ++v)
        cAs[(v + hi * 8) * ROWH + n] = f2bf(h_keep[nt][v]);
    }
    __syncthreads();
  }

  // ---- final projection: p = h_final @ Wph + bp  (tiny: 16x10 per WG) ----
  if (tid < BROWS * 10) {
    const int b = tid / 10, c = tid % 10;
    float s = bp[c];
    for (int k = 0; k < H; ++k)
      s += bf2f(cAs[b * ROWH + k]) * Wph[k * 10 + c];
    out[(wg * BROWS + b) * 10 + c] = s;
  }
}

// ---------------------------------------------------------------------------
extern "C" void kernel_launch(void* const* d_in, const int* in_sizes, int n_in,
                              void* d_out, int out_size, void* d_ws, size_t ws_size,
                              hipStream_t stream) {
  const int*   x    = (const int*)d_in[0];
  const float* emb  = (const float*)d_in[1];
  const float* Wz   = (const float*)d_in[2];
  const float* Wr   = (const float*)d_in[3];
  const float* W    = (const float*)d_in[4];
  const float* Wph  = (const float*)d_in[5];
  const float* bp   = (const float*)d_in[6];
  float*       out  = (float*)d_out;
  unsigned short* wbuf = (unsigned short*)d_ws;   // 3*64*33*512*2 B ~= 6.5 MB

  const int total = 3 * NT * KT * 512;
  gru_prep<<<(total + 255) / 256, 256, 0, stream>>>(Wz, Wr, W, wbuf);
  gru_main<<<16, 512, 0, stream>>>(x, emb, wbuf, Wph, bp, out);
}
